// _NonLocalBlockND_65438121722146
// MI455X (gfx1250) — matrix operand order repaired and checked by reading, not compiled, not run on hardware
//
#include <hip/hip_runtime.h>
#include <hip/hip_bf16.h>

typedef __bf16 bf16;
typedef __attribute__((ext_vector_type(16))) __bf16 v16bf;
typedef __attribute__((ext_vector_type(8)))  __bf16 v8bf;
typedef __attribute__((ext_vector_type(8)))  float  v8f;

static constexpr int BATCH = 4;
static constexpr int CIN   = 256;
static constexpr int CI    = 128;   // inter channels
static constexpr int COUT  = 256;
static constexpr int NSP   = 4096;  // 64*64
static constexpr int KT    = 64;    // flash-attention key tile

union ABfrag { v16bf v; v8bf h[2]; };

static __device__ __forceinline__ v8f vzero8() {
  v8f r;
#pragma unroll
  for (int i = 0; i < 8; ++i) r[i] = 0.0f;
  return r;
}

static __device__ __forceinline__ v8f wmma_bf16(const v16bf& a, const v16bf& b, const v8f& c) {
  // D = A(16x32 bf16) * B(32x16 bf16) + C(16x16 f32)
  return __builtin_amdgcn_wmma_f32_16x16x32_bf16(false, a, false, b, (short)0, c, false, false);
}

// Async 16-byte copy global -> LDS (tracked with ASYNCcnt, no VGPR round-trip).
// VDST carries the wave-relative LDS byte address (low 32 bits of generic ptr).
static __device__ __forceinline__ void async_copy_b128(const void* g, void* l) {
  unsigned lds = (unsigned)(uintptr_t)l;
  asm volatile("global_load_async_to_lds_b128 %0, %1, off"
               :: "v"(lds), "v"(g) : "memory");
}

static __device__ __forceinline__ void wait_async0() {
  asm volatile("s_wait_asynccnt 0x0" ::: "memory");
}

// ---------------------------------------------------------------------------
// x (B,Cin,N) fp32  ->  xT (B,N,Cin) bf16   (write-coalesced)
// ---------------------------------------------------------------------------
__global__ __launch_bounds__(256) void k_cvt_T(const float* __restrict__ x,
                                               bf16* __restrict__ xT) {
  size_t idx = (size_t)blockIdx.x * 256 + threadIdx.x;
  const size_t total = (size_t)BATCH * NSP * CIN;
  if (idx >= total) return;
  int c = (int)(idx % CIN);
  size_t t = idx / CIN;
  int n = (int)(t % NSP);
  int b = (int)(t / NSP);
  xT[idx] = (bf16)x[((size_t)b * CIN + c) * NSP + n];
}

// ---------------------------------------------------------------------------
// Projection GEMM: out[b,o,n] = sum_c w[o,c]*x[b,c,n] + bias[o]
// M=CI(=128, whole-M tile), N-tile=128, K=CIN=256.  8 waves x 16 M-rows.
// ---------------------------------------------------------------------------
__global__ __launch_bounds__(256)
void k_proj(const bf16* __restrict__ xT, const float* __restrict__ w,
            const float* __restrict__ bias, float* __restrict__ out) {
  __shared__ bf16 lw[CI][CIN];  // 64 KB bf16 weight stage
  const int tid = threadIdx.x;
  const int b = blockIdx.y;
  const int n0 = blockIdx.x * 128;
  for (int i = tid; i < CI * CIN; i += 256) lw[i / CIN][i % CIN] = (bf16)w[i];
  __syncthreads();

  const int wv = tid >> 5, lane = tid & 31, ln = lane & 15, hi = lane >> 4;
  const int m0 = wv * 16;

  v8f acc[8];
#pragma unroll
  for (int nt = 0; nt < 8; ++nt) acc[nt] = vzero8();

#pragma unroll
  for (int kc = 0; kc < CIN / 32; ++kc) {
    ABfrag a;
    const bf16* ap = &lw[m0 + ln][kc * 32 + (hi ? 8 : 0)];
    a.h[0] = *(const v8bf*)ap;          // K = base .. base+7
    a.h[1] = *(const v8bf*)(ap + 16);   // K = base+16 .. base+23
#pragma unroll
    for (int nt = 0; nt < 8; ++nt) {
      ABfrag bb;
      const bf16* bp =
          &xT[((size_t)b * NSP + n0 + nt * 16 + ln) * CIN + kc * 32 + (hi ? 8 : 0)];
      bb.h[0] = *(const v8bf*)bp;       // K = base .. base+7
      bb.h[1] = *(const v8bf*)(bp + 16);
      acc[nt] = wmma_bf16(a.v, bb.v, acc[nt]);
    }
  }

  float bv[8];
#pragma unroll
  for (int v = 0; v < 8; ++v) bv[v] = bias[m0 + v + hi * 8];
#pragma unroll
  for (int nt = 0; nt < 8; ++nt)
#pragma unroll
    for (int v = 0; v < 8; ++v) {
      int m = m0 + v + hi * 8;
      int n = n0 + nt * 16 + ln;
      out[((size_t)b * CI + m) * NSP + n] = acc[nt][v] + bv[v];
    }
}

// ---------------------------------------------------------------------------
// Per-channel mean / rstd over (B, N) of src (B,C,N) -> stats[2C]
// ---------------------------------------------------------------------------
__global__ __launch_bounds__(256)
void k_stats(const float* __restrict__ src, float* __restrict__ stats, int C) {
  const int c = blockIdx.x;
  const int tid = threadIdx.x;
  const size_t cnt = (size_t)BATCH * NSP;
  float s = 0.f, s2 = 0.f;
  for (size_t i = tid; i < cnt; i += 256) {
    size_t b = i / NSP, n = i % NSP;
    float v = src[(b * (size_t)C + c) * NSP + n];
    s += v;
    s2 += v * v;
  }
  __shared__ float rs[256], rq[256];
  rs[tid] = s;
  rq[tid] = s2;
  __syncthreads();
  for (int st = 128; st > 0; st >>= 1) {
    if (tid < st) { rs[tid] += rs[tid + st]; rq[tid] += rq[tid + st]; }
    __syncthreads();
  }
  if (tid == 0) {
    float inv = 1.0f / (float)cnt;
    float mean = rs[0] * inv;
    float var = rq[0] * inv - mean * mean;
    stats[2 * c] = mean;
    stats[2 * c + 1] = rsqrtf(var + 1e-5f);
  }
}

// BN apply, same layout (B,C,N) -> bf16
__global__ __launch_bounds__(256)
void k_bn_cvt(const float* __restrict__ raw, const float* __restrict__ stats,
              const float* __restrict__ gamma, const float* __restrict__ beta,
              bf16* __restrict__ out, int C) {
  size_t idx = (size_t)blockIdx.x * 256 + threadIdx.x;
  const size_t total = (size_t)BATCH * C * NSP;
  if (idx >= total) return;
  int c = (int)((idx / NSP) % C);
  float v = raw[idx];
  out[idx] = (bf16)((v - stats[2 * c]) * stats[2 * c + 1] * gamma[c] + beta[c]);
}

// BN apply + transpose: raw (B,C,N) -> out (B,N,C) bf16
__global__ __launch_bounds__(256)
void k_bn_cvt_T(const float* __restrict__ raw, const float* __restrict__ stats,
                const float* __restrict__ gamma, const float* __restrict__ beta,
                bf16* __restrict__ out, int C) {
  size_t idx = (size_t)blockIdx.x * 256 + threadIdx.x;  // (b,n,c) order
  const size_t total = (size_t)BATCH * C * NSP;
  if (idx >= total) return;
  int c = (int)(idx % C);
  size_t t = idx / C;
  int n = (int)(t % NSP);
  int b = (int)(t / NSP);
  float v = raw[((size_t)b * C + c) * NSP + n];
  out[idx] = (bf16)((v - stats[2 * c]) * stats[2 * c + 1] * gamma[c] + beta[c]);
}

// final BN -> fp32 out
__global__ __launch_bounds__(256)
void k_bn_out(const float* __restrict__ z, const float* __restrict__ stats,
              const float* __restrict__ gamma, const float* __restrict__ beta,
              float* __restrict__ out, int C) {
  size_t idx = (size_t)blockIdx.x * 256 + threadIdx.x;
  const size_t total = (size_t)BATCH * C * NSP;
  if (idx >= total) return;
  int c = (int)((idx / NSP) % C);
  out[idx] = (z[idx] - stats[2 * c]) * stats[2 * c + 1] * gamma[c] + beta[c];
}

// ---------------------------------------------------------------------------
// Flash attention: yT[b,q,c] = sum_m softmax_m(theta[:,q].phi[:,m]) * g[c,m]
// th (B,Ci,N) bf16, phT (B,N,Ci) bf16, gg (B,Ci,N) bf16 -> yT (B,N,Ci) bf16
// block = 8 waves, 128 query rows; per wave 16 queries held as A-fragments.
// Key tiles staged in LDS via GLOBAL_LOAD_ASYNC_TO_LDS_B128 (ASYNCcnt path).
// ---------------------------------------------------------------------------
__global__ __launch_bounds__(256)
void k_attn(const bf16* __restrict__ th, const bf16* __restrict__ phT,
            const bf16* __restrict__ gg, bf16* __restrict__ yT) {
  __shared__ bf16 lphT[KT][CI + 8];    // [key][c]   17408 B (row stride 272 B)
  __shared__ bf16 lg[CI][KT + 8];      // [c][key]   18432 B (row stride 144 B)
  __shared__ bf16 pst[8][16][KT + 8];  // per-wave P staging, 18432 B
  const int tid = threadIdx.x;
  const int b = blockIdx.y;
  const int q0 = blockIdx.x * 128;
  const int wv = tid >> 5, lane = tid & 31, ln = lane & 15, hi = lane >> 4;
  const int q = q0 + wv * 16 + ln;

  // theta A-fragments, kept in VGPRs for the whole key loop (K = Ci = 4x32)
  ABfrag at[4];
#pragma unroll
  for (int kc = 0; kc < 4; ++kc) {
#pragma unroll
    for (int i = 0; i < 8; ++i) {
      int c0 = kc * 32 + (hi ? 8 : 0) + i;
      at[kc].v[i]     = th[((size_t)b * CI + c0) * NSP + q];
      at[kc].v[i + 8] = th[((size_t)b * CI + c0 + 16) * NSP + q];
    }
  }

  v8f yacc[8];
#pragma unroll
  for (int ct = 0; ct < 8; ++ct) yacc[ct] = vzero8();
  float mrun[8], lrun[8];
#pragma unroll
  for (int v = 0; v < 8; ++v) { mrun[v] = -1e30f; lrun[v] = 0.f; }

  for (int k0 = 0; k0 < NSP; k0 += KT) {
    __syncthreads();
    // -------- async DMA: phi^T tile (KT x CI) and g tile (CI x KT) to LDS ---
    {
      const bf16* gph = &phT[((size_t)b * NSP + k0) * CI];
#pragma unroll
      for (int i = 0; i < 4; ++i) {          // KT*CI/8 = 1024 16B chunks
        int j = tid + i * 256;
        int row = j >> 4;                    // key row 0..63
        int col = (j & 15) << 3;             // 8-element column chunk
        async_copy_b128(gph + (size_t)row * CI + col, &lphT[row][col]);
      }
      const bf16* ggb = &gg[(size_t)b * CI * NSP + k0];
#pragma unroll
      for (int i = 0; i < 4; ++i) {          // CI*KT/8 = 1024 16B chunks
        int j = tid + i * 256;
        int c = j >> 3;                      // channel row 0..127
        int seg = (j & 7) << 3;              // 8-element key chunk
        async_copy_b128(ggb + (size_t)c * NSP + seg, &lg[c][seg]);
      }
    }
    if (k0 + KT < NSP) {
      __builtin_prefetch(&gg[(size_t)b * CI * NSP + k0 + KT], 0, 1);
      __builtin_prefetch(&phT[((size_t)b * NSP + k0 + KT) * CI], 0, 1);
    }
    wait_async0();
    __syncthreads();

    // S = theta_q^T . phi   (16 x KT)
    v8f s[KT / 16];
#pragma unroll
    for (int nt = 0; nt < KT / 16; ++nt) s[nt] = vzero8();
#pragma unroll
    for (int kc = 0; kc < 4; ++kc) {
#pragma unroll
      for (int nt = 0; nt < KT / 16; ++nt) {
        ABfrag bb;
        const bf16* bp = &lphT[nt * 16 + ln][kc * 32 + (hi ? 8 : 0)];
        bb.h[0] = *(const v8bf*)bp;
        bb.h[1] = *(const v8bf*)(bp + 16);
        s[nt] = wmma_bf16(at[kc].v, bb.v, s[nt]);
      }
    }

    // online softmax (row reductions across 16-lane half groups; wave32)
#pragma unroll
    for (int v = 0; v < 8; ++v) {
      float mt = -1e30f;
#pragma unroll
      for (int nt = 0; nt < KT / 16; ++nt) mt = fmaxf(mt, s[nt][v]);
#pragma unroll
      for (int off = 1; off < 16; off <<= 1) mt = fmaxf(mt, __shfl_xor(mt, off, 32));
      float mnew = fmaxf(mrun[v], mt);
      float scale = __expf(mrun[v] - mnew);
      float lt = 0.f;
#pragma unroll
      for (int nt = 0; nt < KT / 16; ++nt) {
        float p = __expf(s[nt][v] - mnew);
        s[nt][v] = p;
        lt += p;
      }
#pragma unroll
      for (int off = 1; off < 16; off <<= 1) lt += __shfl_xor(lt, off, 32);
      lrun[v] = lrun[v] * scale + lt;
      mrun[v] = mnew;
#pragma unroll
      for (int ct = 0; ct < 8; ++ct) yacc[ct][v] *= scale;
    }

    // stage P (C/D layout) -> LDS -> reload as A-fragments (per-wave region,
    // LDS ops from one wave are in-order, no barrier needed)
#pragma unroll
    for (int nt = 0; nt < KT / 16; ++nt)
#pragma unroll
      for (int v = 0; v < 8; ++v)
        pst[wv][v + hi * 8][nt * 16 + ln] = (bf16)s[nt][v];

    ABfrag ap[KT / 32];
#pragma unroll
    for (int kc = 0; kc < KT / 32; ++kc) {
      const bf16* p = &pst[wv][ln][kc * 32 + (hi ? 8 : 0)];
      ap[kc].h[0] = *(const v8bf*)p;
      ap[kc].h[1] = *(const v8bf*)(p + 16);
    }

    // y += P . g^T  (K = keys)
#pragma unroll
    for (int ct = 0; ct < 8; ++ct)
#pragma unroll
      for (int kc = 0; kc < KT / 32; ++kc) {
        ABfrag bb;
        const bf16* bp = &lg[ct * 16 + ln][kc * 32 + (hi ? 8 : 0)];
        bb.h[0] = *(const v8bf*)bp;
        bb.h[1] = *(const v8bf*)(bp + 16);
        yacc[ct] = wmma_bf16(ap[kc].v, bb.v, yacc[ct]);
      }
  }

#pragma unroll
  for (int ct = 0; ct < 8; ++ct)
#pragma unroll
    for (int v = 0; v < 8; ++v) {
      int row = q0 + wv * 16 + v + hi * 8;
      float val = yacc[ct][v] / lrun[v];
      yT[((size_t)b * NSP + row) * CI + ct * 16 + ln] = (bf16)val;
    }
}

// ---------------------------------------------------------------------------
// W-conv + residual: z[b,o,n] = sum_c Ww[o,c]*yT[b,n,c] + Wb[o] + x[b,o,n]
// M=COUT=256 (2 tiles/wave), N-tile=64, K=CI=128.
// ---------------------------------------------------------------------------
__global__ __launch_bounds__(256)
void k_wconv(const float* __restrict__ Ww, const float* __restrict__ Wb,
             const bf16* __restrict__ yT, const float* __restrict__ x,
             float* __restrict__ z) {
  __shared__ bf16 lw[COUT][CI];  // 64 KB
  const int tid = threadIdx.x;
  const int b = blockIdx.y;
  const int n0 = blockIdx.x * 64;
  for (int i = tid; i < COUT * CI; i += 256) lw[i / CI][i % CI] = (bf16)Ww[i];
  __syncthreads();

  const int wv = tid >> 5, lane = tid & 31, ln = lane & 15, hi = lane >> 4;
  const int m0 = wv * 32;
  v8f acc[2][4];
#pragma unroll
  for (int mi = 0; mi < 2; ++mi)
#pragma unroll
    for (int nt = 0; nt < 4; ++nt) acc[mi][nt] = vzero8();

#pragma unroll
  for (int kc = 0; kc < CI / 32; ++kc) {
    ABfrag a[2];
#pragma unroll
    for (int mi = 0; mi < 2; ++mi) {
      const bf16* apn = &lw[m0 + mi * 16 + ln][kc * 32 + (hi ? 8 : 0)];
      a[mi].h[0] = *(const v8bf*)apn;
      a[mi].h[1] = *(const v8bf*)(apn + 16);
    }
#pragma unroll
    for (int nt = 0; nt < 4; ++nt) {
      ABfrag bb;
      const bf16* bp =
          &yT[((size_t)b * NSP + n0 + nt * 16 + ln) * CI + kc * 32 + (hi ? 8 : 0)];
      bb.h[0] = *(const v8bf*)bp;
      bb.h[1] = *(const v8bf*)(bp + 16);
#pragma unroll
      for (int mi = 0; mi < 2; ++mi) acc[mi][nt] = wmma_bf16(a[mi].v, bb.v, acc[mi][nt]);
    }
  }

#pragma unroll
  for (int mi = 0; mi < 2; ++mi) {
    float wb[8];
#pragma unroll
    for (int v = 0; v < 8; ++v) wb[v] = Wb[m0 + mi * 16 + v + hi * 8];
#pragma unroll
    for (int nt = 0; nt < 4; ++nt)
#pragma unroll
      for (int v = 0; v < 8; ++v) {
        int m = m0 + mi * 16 + v + hi * 8;
        int n = n0 + nt * 16 + ln;
        size_t o = ((size_t)b * COUT + m) * NSP + n;
        z[o] = acc[mi][nt][v] + wb[v] + x[o];
      }
  }
}

// ---------------------------------------------------------------------------
extern "C" void kernel_launch(void* const* d_in, const int* in_sizes, int n_in,
                              void* d_out, int out_size, void* d_ws, size_t ws_size,
                              hipStream_t stream) {
  const float* x        = (const float*)d_in[0];
  const float* g_w      = (const float*)d_in[1];
  const float* g_b      = (const float*)d_in[2];
  const float* g_gamma  = (const float*)d_in[3];
  const float* g_beta   = (const float*)d_in[4];
  const float* th_w     = (const float*)d_in[5];
  const float* th_b     = (const float*)d_in[6];
  const float* th_gamma = (const float*)d_in[7];
  const float* th_beta  = (const float*)d_in[8];
  const float* ph_w     = (const float*)d_in[9];
  const float* ph_b     = (const float*)d_in[10];
  const float* ph_gamma = (const float*)d_in[11];
  const float* ph_beta  = (const float*)d_in[12];
  const float* W_w      = (const float*)d_in[13];
  const float* W_b      = (const float*)d_in[14];
  const float* bn_gamma = (const float*)d_in[15];
  const float* bn_beta  = (const float*)d_in[16];
  float* out = (float*)d_out;

  char* ws = (char*)d_ws;
  const size_t MB = (size_t)1 << 20;
  bf16*  xT     = (bf16*)(ws + 0 * MB);    // 8 MB  (B,N,Cin) bf16
  float* raw_g  = (float*)(ws + 8 * MB);   // 8 MB each, fp32 (B,Ci,N)
  float* raw_th = (float*)(ws + 16 * MB);
  float* raw_ph = (float*)(ws + 24 * MB);
  bf16*  gb     = (bf16*)(ws + 32 * MB);   // 4 MB  (B,Ci,N)
  bf16*  thb    = (bf16*)(ws + 36 * MB);   // 4 MB  (B,Ci,N)
  bf16*  phTn   = (bf16*)(ws + 40 * MB);   // 4 MB  (B,N,Ci)
  bf16*  yTn    = (bf16*)(ws + 44 * MB);   // 4 MB  (B,N,Ci)
  float* z      = (float*)(ws + 8 * MB);   // 16 MB, overlays dead raw_g/raw_th
  float* st_g   = (float*)(ws + 48 * MB);
  float* st_th  = st_g + 2 * CI;
  float* st_ph  = st_th + 2 * CI;
  float* st_z   = st_ph + 2 * CI;

  const int totalX = BATCH * CIN * NSP;  // 4,194,304
  const int totalP = BATCH * CI * NSP;   // 2,097,152
  const int totalZ = BATCH * COUT * NSP; // 4,194,304

  k_cvt_T<<<totalX / 256, 256, 0, stream>>>(x, xT);

  dim3 pg(NSP / 128, BATCH);
  k_proj<<<pg, 256, 0, stream>>>(xT, g_w, g_b, raw_g);
  k_proj<<<pg, 256, 0, stream>>>(xT, th_w, th_b, raw_th);
  k_proj<<<pg, 256, 0, stream>>>(xT, ph_w, ph_b, raw_ph);

  k_stats<<<CI, 256, 0, stream>>>(raw_g, st_g, CI);
  k_stats<<<CI, 256, 0, stream>>>(raw_th, st_th, CI);
  k_stats<<<CI, 256, 0, stream>>>(raw_ph, st_ph, CI);

  k_bn_cvt<<<totalP / 256, 256, 0, stream>>>(raw_g, st_g, g_gamma, g_beta, gb, CI);
  k_bn_cvt<<<totalP / 256, 256, 0, stream>>>(raw_th, st_th, th_gamma, th_beta, thb, CI);
  k_bn_cvt_T<<<totalP / 256, 256, 0, stream>>>(raw_ph, st_ph, ph_gamma, ph_beta, phTn, CI);

  k_attn<<<dim3(NSP / 128, BATCH), 256, 0, stream>>>(thb, phTn, gb, yTn);

  k_wconv<<<dim3(NSP / 64, BATCH), 256, 0, stream>>>(W_w, W_b, yTn, x, z);

  k_stats<<<COUT, 256, 0, stream>>>(z, st_z, COUT);
  k_bn_out<<<totalZ / 256, 256, 0, stream>>>(z, st_z, bn_gamma, bn_beta, out, COUT);
}
